// DeepMarkovModel_39444979647111
// MI455X (gfx1250) — matrix operand order repaired and checked by reading, not compiled, not run on hardware
//
#include <hip/hip_runtime.h>
#include <hip/hip_bf16.h>
#include <stdint.h>

// ---------------------------------------------------------------------------
// Problem constants (from reference)
// ---------------------------------------------------------------------------
#define T_DIM   8192
#define X_DIM   512
#define Z_DIM   512
#define H_DIM   2048
#define IN_DIM  (X_DIM + Z_DIM)   // 1024
#define LOG2PI  1.8378770664093453f

// Try the CDNA5 async global->LDS path (inline asm; flip to 0 if assembler rejects)
#define USE_ASYNC_LDS 1

// ---------------------------------------------------------------------------
// Types for WMMA (CDNA5 gfx1250, wave32)
// ---------------------------------------------------------------------------
typedef __attribute__((ext_vector_type(16))) __bf16 v16bf;
typedef __attribute__((ext_vector_type(8)))  __bf16 v8bf;
typedef __attribute__((ext_vector_type(8)))  float  v8f;

__device__ __forceinline__ unsigned short f2bf(float f) {
    unsigned u = __float_as_uint(f);
    u += 0x7FFFu + ((u >> 16) & 1u);   // round-to-nearest-even
    return (unsigned short)(u >> 16);
}
__device__ __forceinline__ float sigmoidf(float x) { return 1.0f / (1.0f + expf(-x)); }

// ---------------------------------------------------------------------------
// Grid barrier (atomic counter + generation, agent scope, s_sleep spin)
// ---------------------------------------------------------------------------
__device__ __forceinline__ void grid_barrier(unsigned* cnt, unsigned* gen, unsigned nblocks) {
    __syncthreads();
    if (threadIdx.x == 0) {
        __threadfence();
        unsigned g = __hip_atomic_load(gen, __ATOMIC_ACQUIRE, __HIP_MEMORY_SCOPE_AGENT);
        unsigned a = __hip_atomic_fetch_add(cnt, 1u, __ATOMIC_ACQ_REL, __HIP_MEMORY_SCOPE_AGENT);
        if (a == nblocks - 1u) {
            __hip_atomic_store(cnt, 0u, __ATOMIC_RELAXED, __HIP_MEMORY_SCOPE_AGENT);
            __hip_atomic_fetch_add(gen, 1u, __ATOMIC_RELEASE, __HIP_MEMORY_SCOPE_AGENT);
        } else {
            while (__hip_atomic_load(gen, __ATOMIC_ACQUIRE, __HIP_MEMORY_SCOPE_AGENT) == g) {
                __builtin_amdgcn_s_sleep(2);
            }
        }
    }
    __syncthreads();
}

// ---------------------------------------------------------------------------
// Phase 1: sequential inference-net scan, weights resident in LDS.
// 64 blocks x 256 threads. Block b owns Wi1 rows [32b,32b+32), Wi2 rows [16b,16b+16).
// LDS: 32*1024 + 16*2048 + 1024 + 2048 floats = 274,432 bytes (< 320 KB/WGP).
// ---------------------------------------------------------------------------
#define SEQ_NBLK 64
#define SEQ_LDS_BYTES ((32*1024 + 16*2048 + 1024 + 2048) * 4)

__global__ __launch_bounds__(256)
void k_seq_scan(const float* __restrict__ x, const float* __restrict__ eps,
                const float* __restrict__ Wi1, const float* __restrict__ bi1,
                const float* __restrict__ Wi2, const float* __restrict__ bi2,
                float* __restrict__ MUQ, float* __restrict__ LVQ,
                float* __restrict__ gH, unsigned* __restrict__ sync)
{
    extern __shared__ float lds[];
    float* sW1 = lds;                    // 32 x 1024
    float* sW2 = sW1 + 32 * 1024;        // 16 x 2048
    float* sV  = sW2 + 16 * 2048;        // 1024
    float* sH  = sV + 1024;              // 2048

    const int b   = blockIdx.x;
    const int tid = threadIdx.x;
    unsigned* cnt = sync;
    unsigned* gen = sync + 1;

    // Load weight slices into LDS once (float4 vectorized).
    {
        const float4* src1 = (const float4*)(Wi1 + (size_t)b * 32 * 1024);
        for (int i = tid; i < 32 * 1024 / 4; i += 256) ((float4*)sW1)[i] = src1[i];
        const float4* src2 = (const float4*)(Wi2 + (size_t)b * 16 * 2048);
        for (int i = tid; i < 16 * 2048 / 4; i += 256) ((float4*)sW2)[i] = src2[i];
    }
    __syncthreads();

    for (int t = 0; t < T_DIM; ++t) {
        // --- stage v = [x_t ; z_{t-1}] (z recomputed from MUQ/LVQ/eps of t-1)
        for (int i = tid; i < X_DIM; i += 256) sV[i] = x[(size_t)t * X_DIM + i];
        for (int j = tid; j < Z_DIM; j += 256) {
            float zp = 0.0f;
            if (t > 0) {
                size_t idx = (size_t)(t - 1) * Z_DIM + j;
                zp = MUQ[idx] + expf(0.5f * LVQ[idx]) * eps[idx];
            }
            sV[X_DIM + j] = zp;
        }
        __syncthreads();

        // --- matvec1: 32 rows, 8 threads/row x 128 cols (float4), relu -> global h
        {
            const int rl = tid >> 3, part = tid & 7;
            const float4* wr4 = (const float4*)(sW1 + rl * IN_DIM + part * 128);
            const float4* vv4 = (const float4*)(sV + part * 128);
            float acc = 0.0f;
            #pragma unroll 8
            for (int c = 0; c < 32; ++c) {
                float4 wv = wr4[c], xv = vv4[c];
                acc += wv.x * xv.x + wv.y * xv.y + wv.z * xv.z + wv.w * xv.w;
            }
            acc += __shfl_xor(acc, 1);
            acc += __shfl_xor(acc, 2);
            acc += __shfl_xor(acc, 4);
            if (part == 0) {
                int rg = b * 32 + rl;
                float h = acc + bi1[rg];
                gH[rg] = h > 0.0f ? h : 0.0f;
            }
        }
        grid_barrier(cnt, gen, SEQ_NBLK);

        // --- gather full h (8 KB) into LDS
#if USE_ASYNC_LDS
        {
            // each thread: 8 floats = 2 x B128 async global->LDS transfers
            unsigned loff = (unsigned)(size_t)(void*)(sH + tid * 8);
            unsigned long long ga = (unsigned long long)(gH + tid * 8);
            asm volatile(
                "global_load_async_to_lds_b128 %0, %1, off\n\t"
                "global_load_async_to_lds_b128 %0, %1, off offset:16"
                :: "v"(loff), "v"(ga) : "memory");
            asm volatile("s_wait_asynccnt 0x0" ::: "memory");
        }
        __syncthreads();
#else
        for (int i = tid; i < H_DIM; i += 256) sH[i] = gH[i];
        __syncthreads();
#endif

        // --- matvec2: 16 rows, 16 threads/row x 128 cols (float4) -> mu_q/logvar_q
        {
            const int rl = tid >> 4, part = tid & 15;
            const float4* wr4 = (const float4*)(sW2 + rl * H_DIM + part * 128);
            const float4* hh4 = (const float4*)(sH + part * 128);
            float acc = 0.0f;
            #pragma unroll 8
            for (int c = 0; c < 32; ++c) {
                float4 wv = wr4[c], hv = hh4[c];
                acc += wv.x * hv.x + wv.y * hv.y + wv.z * hv.z + wv.w * hv.w;
            }
            acc += __shfl_xor(acc, 1);
            acc += __shfl_xor(acc, 2);
            acc += __shfl_xor(acc, 4);
            acc += __shfl_xor(acc, 8);
            if (part == 0) {
                int n = b * 16 + rl;
                float q = acc + bi2[n];
                if (n < Z_DIM) MUQ[(size_t)t * Z_DIM + n]           = q;
                else           LVQ[(size_t)t * Z_DIM + (n - Z_DIM)] = q;
            }
        }
        grid_barrier(cnt, gen, SEQ_NBLK);
    }
}

// ---------------------------------------------------------------------------
// fp32 -> bf16 conversion (grid-stride)
// ---------------------------------------------------------------------------
__global__ __launch_bounds__(256)
void k_cvt_bf16(const float* __restrict__ src, unsigned short* __restrict__ dst, size_t n)
{
    for (size_t i = (size_t)blockIdx.x * 256 + threadIdx.x; i < n; i += (size_t)gridDim.x * 256)
        dst[i] = f2bf(src[i]);
}

// Build Z (z_t) and Zprev (z_{t-1}, row 0 = 0) in bf16 from MUQ/LVQ/eps.
__global__ __launch_bounds__(256)
void k_build_z(const float* __restrict__ MUQ, const float* __restrict__ LVQ,
               const float* __restrict__ eps,
               unsigned short* __restrict__ Zbf, unsigned short* __restrict__ Zpbf)
{
    const size_t n = (size_t)T_DIM * Z_DIM;
    for (size_t i = (size_t)blockIdx.x * 256 + threadIdx.x; i < n; i += (size_t)gridDim.x * 256) {
        size_t t = i >> 9;                 // /512
        float z = MUQ[i] + expf(0.5f * LVQ[i]) * eps[i];
        Zbf[i] = f2bf(z);
        if (t + 1 < T_DIM) Zpbf[i + Z_DIM] = f2bf(z);
        if (i < Z_DIM)     Zpbf[i]         = 0;   // bf16(0)
    }
}

// ---------------------------------------------------------------------------
// WMMA bf16 GEMM: C(MxN) = A(MxK) * B(NxK)^T   (+bias, +relu/bf16 variants)
// Block 256 thr = 8 waves (2 in M x 4 in N); wave tile 64x32 (4x2 WMMA tiles);
// block tile 128x128. Software-pipelined 2-deep over K (K/32 even: 512, 2048).
// EPI: 0 = fp32 out; 1 = fp32 out + bias; 2 = bias + relu + bf16 out.
// ---------------------------------------------------------------------------
struct Frags { v16bf a[4]; v16bf b[2]; };

__device__ __forceinline__ void load_frags(Frags& f,
        const __bf16* __restrict__ A, const __bf16* __restrict__ B,
        int wrow, int wcol, int K, int k0, int half, int l15, bool pf)
{
    #pragma unroll
    for (int i = 0; i < 4; ++i) {
        const __bf16* ap = A + (size_t)(wrow + i * 16 + l15) * K + k0;
        f.a[i] = *(const v16bf*)(ap + half * 16);
        if (pf) __builtin_prefetch(ap + 64, 0, 1);   // unconditional, speculative
    }
    #pragma unroll
    for (int j = 0; j < 2; ++j)
        f.b[j] = *(const v16bf*)(B + (size_t)(wcol + j * 16 + l15) * K + k0 + half * 16);
}

__device__ __forceinline__ void do_wmma(v8f acc[4][2], const Frags& f)
{
    #pragma unroll
    for (int i = 0; i < 4; ++i)
        #pragma unroll
        for (int j = 0; j < 2; ++j)
            acc[i][j] = __builtin_amdgcn_wmma_f32_16x16x32_bf16(
                false, f.a[i], false, f.b[j], (short)0, acc[i][j], false, false);
}

template <int EPI>
__global__ __launch_bounds__(256)
void k_gemm_bf16(const unsigned short* __restrict__ Araw,
                 const unsigned short* __restrict__ Braw,
                 const float* __restrict__ bias,
                 void* __restrict__ Cout, int M, int N, int K)
{
    const __bf16* A = (const __bf16*)Araw;
    const __bf16* B = (const __bf16*)Braw;

    const int lane = threadIdx.x & 31;
    const int wave = threadIdx.x >> 5;
    const int wm   = wave & 1;
    const int wn   = wave >> 1;
    const int half = lane >> 4;
    const int l15  = lane & 15;

    const int wrow = blockIdx.y * 128 + wm * 64;
    const int wcol = blockIdx.x * 128 + wn * 32;

    v8f acc[4][2] = {};
    Frags f0, f1;

    // prologue
    load_frags(f0, A, B, wrow, wcol, K, 0, half, l15, false);
    int k0 = 0;
    for (; k0 + 64 < K; k0 += 64) {
        load_frags(f1, A, B, wrow, wcol, K, k0 + 32, half, l15, true);
        do_wmma(acc, f0);
        load_frags(f0, A, B, wrow, wcol, K, k0 + 64, half, l15, true);
        do_wmma(acc, f1);
    }
    load_frags(f1, A, B, wrow, wcol, K, K - 32, half, l15, false);
    do_wmma(acc, f0);
    do_wmma(acc, f1);

    // Epilogue. C/D layout: VGPR r -> row (half*8 + r), lane l15 -> col.
    #pragma unroll
    for (int j = 0; j < 2; ++j) {
        const int col = wcol + j * 16 + l15;
        const float bv = (EPI >= 1) ? bias[col] : 0.0f;
        #pragma unroll
        for (int i = 0; i < 4; ++i) {
            const int rbase = wrow + i * 16 + half * 8;
            #pragma unroll
            for (int r = 0; r < 8; ++r) {
                float v = acc[i][j][r] + bv;
                if (EPI == 2) {
                    v = v > 0.0f ? v : 0.0f;
                    ((unsigned short*)Cout)[(size_t)(rbase + r) * N + col] = f2bf(v);
                } else {
                    ((float*)Cout)[(size_t)(rbase + r) * N + col] = v;
                }
            }
        }
    }
}

// ---------------------------------------------------------------------------
// LSTM pointwise (f-gate dead since c0 == 0): h = sig(go)*tanh(sig(gi)*tanh(gg))
// ---------------------------------------------------------------------------
__global__ __launch_bounds__(256)
void k_lstm_h(const float* __restrict__ Gi, const float* __restrict__ Gg,
              const float* __restrict__ Go,
              const float* __restrict__ bih, const float* __restrict__ bhh,
              unsigned short* __restrict__ hbf)
{
    const size_t n = (size_t)T_DIM * H_DIM;
    for (size_t i = (size_t)blockIdx.x * 256 + threadIdx.x; i < n; i += (size_t)gridDim.x * 256) {
        int j = (int)(i & (H_DIM - 1));
        float gi = Gi[i] + bih[j]             + bhh[j];
        float gg = Gg[i] + bih[2 * H_DIM + j] + bhh[2 * H_DIM + j];   // offset 4096
        float go = Go[i] + bih[3 * H_DIM + j] + bhh[3 * H_DIM + j];   // offset 6144
        float c = sigmoidf(gi) * tanhf(gg);
        float h = sigmoidf(go) * tanhf(c);
        hbf[i] = f2bf(h);
    }
}

// ---------------------------------------------------------------------------
// Fused KLD + NLL reduction over (t, j) in T x 512
// ---------------------------------------------------------------------------
__global__ __launch_bounds__(256)
void k_reduce(const float* __restrict__ x, const float* __restrict__ MUQ,
              const float* __restrict__ LVQ, const float* __restrict__ P,
              const float* __restrict__ E, float* __restrict__ accs)
{
    float kld = 0.0f, nll = 0.0f;
    const size_t n = (size_t)T_DIM * Z_DIM;
    for (size_t i = (size_t)blockIdx.x * 256 + threadIdx.x; i < n; i += (size_t)gridDim.x * 256) {
        size_t t = i >> 9;
        int    j = (int)(i & 511);
        float lvq = LVQ[i], muq = MUQ[i];
        float mup = 0.0f, lvp = 0.0f;
        if (t != 0) {
            mup = P[t * 1024 + j];
            lvp = P[t * 1024 + 512 + j];
        }
        float dmu = muq - mup;
        kld += 0.5f * (lvp - lvq + expf(lvq - lvp) + dmu * dmu * expf(-lvp) - 1.0f);

        float mux = E[t * 1024 + j];
        float lvx = E[t * 1024 + 512 + j];
        float dx  = x[i] - mux;
        nll += 0.5f * (LOG2PI + lvx + dx * dx * expf(-lvx));
    }
    #pragma unroll
    for (int m = 16; m > 0; m >>= 1) { kld += __shfl_xor(kld, m); nll += __shfl_xor(nll, m); }
    __shared__ float sk[8], sn[8];
    const int wave = threadIdx.x >> 5, lane = threadIdx.x & 31;
    if (lane == 0) { sk[wave] = kld; sn[wave] = nll; }
    __syncthreads();
    if (wave == 0) {
        kld = (lane < 8) ? sk[lane] : 0.0f;
        nll = (lane < 8) ? sn[lane] : 0.0f;
        #pragma unroll
        for (int m = 4; m > 0; m >>= 1) { kld += __shfl_xor(kld, m); nll += __shfl_xor(nll, m); }
        if (lane == 0) { atomicAdd(accs, kld); atomicAdd(accs + 1, nll); }
    }
}

__global__ void k_writeout(const float* __restrict__ accs, float* __restrict__ out)
{
    float kld = accs[0], nll = accs[1];
    out[0] = kld + nll;   // kld_total - ll_total
    out[1] = kld;
    out[2] = nll;         // -ll_total
}

// ---------------------------------------------------------------------------
// Host orchestration
// ---------------------------------------------------------------------------
extern "C" void kernel_launch(void* const* d_in, const int* in_sizes, int n_in,
                              void* d_out, int out_size, void* d_ws, size_t ws_size,
                              hipStream_t stream)
{
    (void)in_sizes; (void)n_in; (void)out_size; (void)ws_size;

    const float* x    = (const float*)d_in[0];
    const float* eps  = (const float*)d_in[1];
    const float* Wi1  = (const float*)d_in[2];
    const float* bi1  = (const float*)d_in[3];
    const float* Wi2  = (const float*)d_in[4];
    const float* bi2  = (const float*)d_in[5];
    const float* We1  = (const float*)d_in[6];
    const float* be1  = (const float*)d_in[7];
    const float* We2  = (const float*)d_in[8];
    const float* be2  = (const float*)d_in[9];
    const float* Wih  = (const float*)d_in[10];
    /* Whh = d_in[11] unused (multiplies h0 == 0) */
    const float* bih  = (const float*)d_in[12];
    const float* bhh  = (const float*)d_in[13];
    const float* Wt   = (const float*)d_in[14];
    const float* bt   = (const float*)d_in[15];
    float* out = (float*)d_out;

    // ---- workspace layout (256B aligned bump allocator) ----
    uint8_t* w = (uint8_t*)d_ws;
    size_t off = 0;
    auto take = [&](size_t bytes) -> void* {
        void* p = w + off;
        off = (off + bytes + 255) & ~(size_t)255;
        return p;
    };
    unsigned*       sync  = (unsigned*)      take(2 * sizeof(unsigned));
    float*          accs  = (float*)         take(2 * sizeof(float));
    float*          gH    = (float*)         take((size_t)H_DIM * 4);
    float*          MUQ   = (float*)         take((size_t)T_DIM * Z_DIM * 4);
    float*          LVQ   = (float*)         take((size_t)T_DIM * Z_DIM * 4);
    unsigned short* Zbf   = (unsigned short*)take((size_t)T_DIM * Z_DIM * 2);
    unsigned short* Zpbf  = (unsigned short*)take((size_t)T_DIM * Z_DIM * 2);
    unsigned short* Wihbf = (unsigned short*)take((size_t)4 * H_DIM * Z_DIM * 2);
    unsigned short* Wtbf  = (unsigned short*)take((size_t)1024 * H_DIM * 2);
    unsigned short* We1bf = (unsigned short*)take((size_t)H_DIM * Z_DIM * 2);
    unsigned short* We2bf = (unsigned short*)take((size_t)1024 * H_DIM * 2);
    unsigned short* hbf   = (unsigned short*)take((size_t)T_DIM * H_DIM * 2);
    float*          P     = (float*)         take((size_t)T_DIM * 1024 * 4);
    float*          Gi    = (float*)         take((size_t)T_DIM * H_DIM * 4);
    float*          Gg    = (float*)         take((size_t)T_DIM * H_DIM * 4);
    float*          Go    = (float*)         take((size_t)T_DIM * H_DIM * 4);
    // Aliases (G buffers are dead after k_lstm_h):
    unsigned short* HEbf  = (unsigned short*)Gi;   // T x 2048 bf16 (32MB <= 64MB)
    float*          E     = (float*)Gg;            // T x 1024 fp32 (32MB <= 64MB)

    // ---- reset sync + accumulators (capture-safe memset nodes) ----
    hipMemsetAsync(sync, 0, 2 * sizeof(unsigned), stream);
    hipMemsetAsync(accs, 0, 2 * sizeof(float), stream);

    // ---- Phase 1: sequential scan with LDS-resident weights ----
    hipFuncSetAttribute(reinterpret_cast<const void*>(k_seq_scan),
                        hipFuncAttributeMaxDynamicSharedMemorySize, SEQ_LDS_BYTES);
    k_seq_scan<<<SEQ_NBLK, 256, SEQ_LDS_BYTES, stream>>>(
        x, eps, Wi1, bi1, Wi2, bi2, MUQ, LVQ, gH, sync);

    // ---- Phase 2 prep: bf16 conversions ----
    k_build_z<<<4096, 256, 0, stream>>>(MUQ, LVQ, eps, Zbf, Zpbf);
    k_cvt_bf16<<<4096, 256, 0, stream>>>(Wih, Wihbf, (size_t)4 * H_DIM * Z_DIM);
    k_cvt_bf16<<<2048, 256, 0, stream>>>(Wt,  Wtbf,  (size_t)1024 * H_DIM);
    k_cvt_bf16<<<1024, 256, 0, stream>>>(We1, We1bf, (size_t)H_DIM * Z_DIM);
    k_cvt_bf16<<<2048, 256, 0, stream>>>(We2, We2bf, (size_t)1024 * H_DIM);

    // ---- Phase 2: batched WMMA GEMMs ----
    // gates (i, g, o only; f-gate is dead): T x 2048 each, K = 512
    {
        dim3 grid(H_DIM / 128, T_DIM / 128);
        k_gemm_bf16<0><<<grid, 256, 0, stream>>>(Zpbf, Wihbf,                         nullptr, Gi, T_DIM, H_DIM, Z_DIM);
        k_gemm_bf16<0><<<grid, 256, 0, stream>>>(Zpbf, Wihbf + (size_t)2*H_DIM*Z_DIM, nullptr, Gg, T_DIM, H_DIM, Z_DIM);
        k_gemm_bf16<0><<<grid, 256, 0, stream>>>(Zpbf, Wihbf + (size_t)3*H_DIM*Z_DIM, nullptr, Go, T_DIM, H_DIM, Z_DIM);
    }
    k_lstm_h<<<8192, 256, 0, stream>>>(Gi, Gg, Go, bih, bhh, hbf);
    // P = h @ Wt^T + bt : T x 1024, K = 2048
    {
        dim3 grid(1024 / 128, T_DIM / 128);
        k_gemm_bf16<1><<<grid, 256, 0, stream>>>(hbf, Wtbf, bt, P, T_DIM, 1024, H_DIM);
    }
    // He = relu(Z @ We1^T + be1) : T x 2048 bf16, K = 512
    {
        dim3 grid(H_DIM / 128, T_DIM / 128);
        k_gemm_bf16<2><<<grid, 256, 0, stream>>>(Zbf, We1bf, be1, HEbf, T_DIM, H_DIM, Z_DIM);
    }
    // E = He @ We2^T + be2 : T x 1024, K = 2048
    {
        dim3 grid(1024 / 128, T_DIM / 128);
        k_gemm_bf16<1><<<grid, 256, 0, stream>>>(HEbf, We2bf, be2, E, T_DIM, 1024, H_DIM);
    }

    // ---- Phase 3: fused KLD/NLL reduction + output ----
    k_reduce<<<2048, 256, 0, stream>>>(x, MUQ, LVQ, P, E, accs);
    k_writeout<<<1, 1, 0, stream>>>(accs, out);
}
